// SVM_13846974562943
// MI455X (gfx1250) — matrix operand order repaired and checked by reading, not compiled, not run on hardware
//
#include <hip/hip_runtime.h>

#define I_SIZE 64
#define H_SIZE 128
#define O_SIZE 64
#define BATCH  64
#define TLEN   2048
#define G3H    384   // 3*H_SIZE

typedef _Float16 h8_t  __attribute__((ext_vector_type(8)));
typedef _Float16 v16h  __attribute__((ext_vector_type(16)));
typedef float    v8f   __attribute__((ext_vector_type(8)));

// ---------------------------------------------------------------------------
// WMMA wrapper: D = A(16x32 f16) * B(32x16 f16) + C(16x16 f32)
// ---------------------------------------------------------------------------
__device__ __forceinline__ v8f wmma_f16(v16h a, v16h b, v8f c) {
    return __builtin_amdgcn_wmma_f32_16x16x32_f16(
        /*neg_a=*/false, a, /*neg_b=*/false, b,
        /*c_mod=*/(short)0, c, /*reuse_a=*/false, /*reuse_b=*/false);
}

// A-fragment from a row-major f16 tile. `row` points at this lane's M-row.
// ISA layout (16-bit A 16x32): lane<16 -> K = kb+{0..7, 16..23};
//                              lane>=16 -> K = kb+{8..15, 24..31}. hs = lane>>4.
__device__ __forceinline__ v16h load_a_f16(const _Float16* row, int kb, int hs) {
    h8_t lo = *(const h8_t*)(row + kb + 8 * hs);
    h8_t hi = *(const h8_t*)(row + kb + 16 + 8 * hs);
    return __builtin_shufflevector(lo, hi, 0,1,2,3,4,5,6,7,8,9,10,11,12,13,14,15);
}

// Same A-fragment but converting from fp32 source on the fly.
__device__ __forceinline__ v16h load_a_f32(const float* row, int kb, int hs) {
    const float* p0 = row + kb + 8 * hs;
    const float* p1 = row + kb + 16 + 8 * hs;
    v16h a;
    #pragma unroll
    for (int e = 0; e < 8; ++e) { a[e] = (_Float16)p0[e]; a[8 + e] = (_Float16)p1[e]; }
    return a;
}

// B-fragment for GEMM  D[m,n] = sum_k A[m,k] * W[n,k]  from a pre-converted
// f16 weight (row-major [N,Kd]): two 16B loads per lane, K = kb+{0..7,16..23} / kb+{8..15,24..31}.
// B[k][n] = W[n*Kd + k]; lane&15 -> column n0+(lane&15); lane>>4 -> K half.
__device__ __forceinline__ v16h load_b_w16(const _Float16* W, int Kd, int n0, int kb, int lane) {
    return load_a_f16(W + (size_t)(n0 + (lane & 15)) * Kd, kb, lane >> 4);
}

__device__ __forceinline__ float sigmoidf_(float x) { return 1.0f / (1.0f + __expf(-x)); }

// CDNA5 async copy: global -> LDS, 16B per lane, tracked with ASYNCcnt.
__device__ __forceinline__ void async_copy_b128(unsigned lds_addr, unsigned long long gaddr) {
    asm volatile("global_load_async_to_lds_b128 %0, %1, off"
                 :: "v"(lds_addr), "v"(gaddr) : "memory");
}
__device__ __forceinline__ void wait_asynccnt0() {
    asm volatile("s_wait_asynccnt 0x0" ::: "memory");
}

// ---------------------------------------------------------------------------
// Prep: fp32 -> f16 weight conversion (run once; keeps GEMM kernels cvt-free)
// ---------------------------------------------------------------------------
__global__ void cvt_f16_kernel(const float* __restrict__ src, _Float16* __restrict__ dst, int n) {
    int i = blockIdx.x * 256 + threadIdx.x;
    if (i < n) dst[i] = (_Float16)src[i];
}

// ---------------------------------------------------------------------------
// Kernel 1: gx[t,b,0:384] = x[b,t,:] @ W_ih^T + b_ih     (M=B*T, N=384, K=64)
// Stored time-major so the recurrence reads contiguous 24KB slabs per step.
// 1 block = 16 GEMM rows; wave w handles N-tiles {w, w+8, w+16}.
// ---------------------------------------------------------------------------
__global__ void __launch_bounds__(256) gru_gx_kernel(
        const float* __restrict__ x, const _Float16* __restrict__ W_ih16,
        const float* __restrict__ b_ih, float* __restrict__ gx) {
    const int tid = threadIdx.x;
    const int w = tid >> 5, lane = tid & 31, hs = lane >> 4;
    const int Mbase = blockIdx.x * 16;

    const float* xrow = x + (size_t)(Mbase + (lane & 15)) * I_SIZE;  // row = b*T + t
    v16h a0 = load_a_f32(xrow, 0, hs);
    v16h a1 = load_a_f32(xrow, 32, hs);

    #pragma unroll
    for (int i = 0; i < 3; ++i) {
        const int n0 = w * 16 + 128 * i;
        v16h b0 = load_b_w16(W_ih16, I_SIZE, n0, 0, lane);
        v16h b1 = load_b_w16(W_ih16, I_SIZE, n0, 32, lane);
        v8f acc = {};
        acc = wmma_f16(a0, b0, acc);
        acc = wmma_f16(a1, b1, acc);
        const float bias = b_ih[n0 + (lane & 15)];
        #pragma unroll
        for (int j = 0; j < 8; ++j) {
            const int m = j + 8 * hs;
            const int row = Mbase + m;            // = b*TLEN + t
            const int bb = row >> 11;             // / TLEN
            const int tt = row & (TLEN - 1);
            gx[((size_t)tt * BATCH + bb) * G3H + n0 + (lane & 15)] = acc[j] + bias;
        }
    }
}

// ---------------------------------------------------------------------------
// Kernel 2: the sequential GRU scan. 4 blocks, one per 16-row batch group.
// W_hh B-fragments live in registers (12 x v16h per wave, constant over T).
// h (16x128 f16) double-buffered in LDS (padded stride, bank-conflict free).
// gx slabs (24KB/step) stream into a double-buffered LDS ring with
// global_load_async_to_lds_b128, overlapped with the 12 WMMAs + gate math;
// the per-step barrier doubles as the async fence (each wave waits its own
// ASYNCcnt to 0 first, so after s_barrier the whole slab is visible).
// ---------------------------------------------------------------------------
__global__ void __launch_bounds__(256) gru_recur_kernel(
        const float* __restrict__ gx, const _Float16* __restrict__ W_hh16,
        const float* __restrict__ b_hh, _Float16* __restrict__ hseq) {
    constexpr int LDH = 136;                      // padded h row stride (halves)
    constexpr int LDG = 388;                      // padded gx row stride (floats)
    __shared__ __align__(16) _Float16 hb[2][16 * LDH];
    __shared__ __align__(16) float    gbuf[2][16 * LDG];

    const int tid = threadIdx.x;
    const int w = tid >> 5, lane = tid & 31, hs = lane >> 4;
    const int bbase = blockIdx.x * 16;            // batch rows of this group
    const int ncol = w * 16 + (lane & 15);        // this lane's H-column

    // Per-thread async-copy chunk map: 16 rows x 96 b128-chunks, 6 per thread.
    int crow[6], ccol[6];
    #pragma unroll
    for (int k = 0; k < 6; ++k) {
        const int c = tid + k * 256;
        crow[k] = c / 96;
        ccol[k] = (c - crow[k] * 96) * 4;
    }

    // Preload recurrent weights: gate g (r,z,n), K-chunk c.
    v16h B[3][4];
    #pragma unroll
    for (int g = 0; g < 3; ++g)
        #pragma unroll
        for (int c = 0; c < 4; ++c)
            B[g][c] = load_b_w16(W_hh16, H_SIZE, g * H_SIZE + w * 16, c * 32, lane);
    const float bhr = b_hh[ncol];
    const float bhz = b_hh[H_SIZE + ncol];
    const float bhn = b_hh[2 * H_SIZE + ncol];

    for (int k = tid; k < 2 * 16 * LDH; k += 256) ((_Float16*)hb)[k] = (_Float16)0.0f;

    // Prologue: stream slab 0 into gbuf[0].
    {
        const float* slab0 = gx + (size_t)bbase * G3H;
        #pragma unroll
        for (int k = 0; k < 6; ++k)
            async_copy_b128((unsigned)(uintptr_t)&gbuf[0][crow[k] * LDG + ccol[k]],
                            (unsigned long long)(uintptr_t)(slab0 + (tid + k * 256) * 4));
    }
    wait_asynccnt0();
    __syncthreads();

    v8f hprev = {};                               // wave w's own h columns
    int cur = 0;
    for (int t = 0; t < TLEN; ++t) {
        // Stream next slab while we compute this step.
        if (t + 1 < TLEN) {
            const float* slab = gx + ((size_t)(t + 1) * BATCH + bbase) * G3H;
            float* dst = gbuf[(t + 1) & 1];
            #pragma unroll
            for (int k = 0; k < 6; ++k)
                async_copy_b128((unsigned)(uintptr_t)&dst[crow[k] * LDG + ccol[k]],
                                (unsigned long long)(uintptr_t)(slab + (tid + k * 256) * 4));
        }
        if (t + 4 < TLEN) {                       // warm L2 a few steps ahead
            const float* pf = gx + ((size_t)(t + 4) * BATCH + bbase) * G3H;
            __builtin_prefetch(pf + tid * 24, 0, 1);
        }

        const _Float16* arow = &hb[cur][(lane & 15) * LDH];
        v16h a0 = load_a_f16(arow, 0, hs);
        v16h a1 = load_a_f16(arow, 32, hs);
        v16h a2 = load_a_f16(arow, 64, hs);
        v16h a3 = load_a_f16(arow, 96, hs);

        v8f accr = {}, accz = {}, accn = {};
        accr = wmma_f16(a0, B[0][0], accr); accr = wmma_f16(a1, B[0][1], accr);
        accr = wmma_f16(a2, B[0][2], accr); accr = wmma_f16(a3, B[0][3], accr);
        accz = wmma_f16(a0, B[1][0], accz); accz = wmma_f16(a1, B[1][1], accz);
        accz = wmma_f16(a2, B[1][2], accz); accz = wmma_f16(a3, B[1][3], accz);
        accn = wmma_f16(a0, B[2][0], accn); accn = wmma_f16(a1, B[2][1], accn);
        accn = wmma_f16(a2, B[2][2], accn); accn = wmma_f16(a3, B[2][3], accn);

        const float* gt = &gbuf[t & 1][ncol];
        v8f hnew = {};
        #pragma unroll
        for (int j = 0; j < 8; ++j) {
            const int m = j + 8 * hs;
            const float* g = gt + m * LDG;
            const float r  = sigmoidf_(g[0]          + accr[j] + bhr);
            const float z  = sigmoidf_(g[H_SIZE]     + accz[j] + bhz);
            const float nn = tanhf    (g[2 * H_SIZE] + r * (accn[j] + bhn));
            const float h  = (1.0f - z) * nn + z * hprev[j];
            hnew[j] = h;
            hb[cur ^ 1][m * LDH + ncol] = (_Float16)h;
            hseq[((size_t)t * BATCH + bbase + m) * H_SIZE + ncol] = (_Float16)h;
        }
        hprev = hnew;
        wait_asynccnt0();                         // own slab-(t+1) copies done
        __syncthreads();                          // slab + hb[cur^1] visible to all
        cur ^= 1;
    }
}

// ---------------------------------------------------------------------------
// Kernel 3: out[b,t,:] = h[t,b,:] @ W_out^T + b_out   (M=T*B, N=64, K=128)
// 1 block = 2 M-tiles x 4 N-tiles (8 waves, one output tile each).
// ---------------------------------------------------------------------------
__global__ void __launch_bounds__(256) gru_out_kernel(
        const _Float16* __restrict__ hseq, const _Float16* __restrict__ W_out16,
        const float* __restrict__ b_out, float* __restrict__ out) {
    const int tid = threadIdx.x;
    const int w = tid >> 5, lane = tid & 31, hs = lane >> 4;
    const int Mbase = (blockIdx.x * 2 + (w >> 2)) * 16;   // rows in [t,b] order
    const int n0 = (w & 3) * 16;

    const _Float16* arow = hseq + (size_t)(Mbase + (lane & 15)) * H_SIZE;
    v8f acc = {};
    #pragma unroll
    for (int c = 0; c < 4; ++c) {
        v16h a = load_a_f16(arow, c * 32, hs);
        v16h b = load_b_w16(W_out16, H_SIZE, n0, c * 32, lane);
        acc = wmma_f16(a, b, acc);
    }
    const float bias = b_out[n0 + (lane & 15)];
    #pragma unroll
    for (int j = 0; j < 8; ++j) {
        const int m = j + 8 * hs;
        const int row = Mbase + m;                // = t*BATCH + b
        const int bb = row & (BATCH - 1);
        const int tt = row >> 6;
        out[((size_t)bb * TLEN + tt) * O_SIZE + n0 + (lane & 15)] = acc[j] + bias;
    }
}

// ---------------------------------------------------------------------------
extern "C" void kernel_launch(void* const* d_in, const int* in_sizes, int n_in,
                              void* d_out, int out_size, void* d_ws, size_t ws_size,
                              hipStream_t stream) {
    const float* x     = (const float*)d_in[0];
    const float* W_ih  = (const float*)d_in[1];
    const float* W_hh  = (const float*)d_in[2];
    const float* b_ih  = (const float*)d_in[3];
    const float* b_hh  = (const float*)d_in[4];
    const float* W_out = (const float*)d_in[5];
    const float* b_out = (const float*)d_in[6];
    float* out = (float*)d_out;

    // Workspace layout:
    //   gx fp32 [T*B*384] | hseq f16 [T*B*128] | W_ih16 | W_hh16 | W_out16
    char* ws = (char*)d_ws;
    float*    gx     = (float*)ws;            ws += (size_t)TLEN * BATCH * G3H * sizeof(float);
    _Float16* hseq   = (_Float16*)ws;         ws += (size_t)TLEN * BATCH * H_SIZE * sizeof(_Float16);
    _Float16* wih16  = (_Float16*)ws;         ws += (size_t)G3H * I_SIZE * sizeof(_Float16);
    _Float16* whh16  = (_Float16*)ws;         ws += (size_t)G3H * H_SIZE * sizeof(_Float16);
    _Float16* wout16 = (_Float16*)ws;

    const int n_ih  = G3H * I_SIZE;           // 24576
    const int n_hh  = G3H * H_SIZE;           // 49152
    const int n_out = O_SIZE * H_SIZE;        // 8192
    cvt_f16_kernel<<<(n_ih  + 255) / 256, 256, 0, stream>>>(W_ih,  wih16,  n_ih);
    cvt_f16_kernel<<<(n_hh  + 255) / 256, 256, 0, stream>>>(W_hh,  whh16,  n_hh);
    cvt_f16_kernel<<<(n_out + 255) / 256, 256, 0, stream>>>(W_out, wout16, n_out);

    gru_gx_kernel   <<<(BATCH * TLEN) / 16, 256, 0, stream>>>(x, wih16, b_ih, gx);
    gru_recur_kernel<<<BATCH / 16,          256, 0, stream>>>(gx, whh16, b_hh, hseq);
    gru_out_kernel  <<<(TLEN * BATCH) / 32, 256, 0, stream>>>(hseq, wout16, b_out, out);
}
